// CrossAttention_27530740367910
// MI455X (gfx1250) — matrix operand order repaired and checked by reading, not compiled, not run on hardware
//
#include <hip/hip_runtime.h>

// ---------------------------------------------------------------------------
// Fused multi-model cross attention for MI455X (gfx1250, wave32, WMMA bf16)
//   q = feat @ Wq^T + bq ; v = feat @ Wv^T + bv   (k == q per reference bug)
//   scores[b,i,j] = <q_bi, q_bj> - 0.5*(pv[b,i]+pv[b,j]); softmax; z = attn@v
// Pipelined: async weight DMA (global->LDS) + in-register feature prefetch
// double-buffered against WMMA compute; qv output tile aliases dead staging LDS.
// ---------------------------------------------------------------------------

typedef __attribute__((ext_vector_type(16))) __bf16 v16bf;
typedef __attribute__((ext_vector_type(8)))  float  v8f;

#define NMODELS 5
#define BATCH   16384
#define IN_DIM  1024
#define QDIM    512
#define COLS    1024            // 0..511 = Q columns, 512..1023 = V columns
#define BT      16              // batch elements per workgroup
#define ROWS    (NMODELS * BT)  // 80 GEMM rows per workgroup
#define KC      32              // K chunk (one WMMA K-step)
#define NKC     (IN_DIM / KC)   // 32 chunks
#define WSTR    20              // staging row stride in dwords (16 + 4 pad, 16B aligned)
#define QV_STR  (COLS + 8)      // qv LDS row stride in bf16 elems (pad)
#define UW      0.5f

// LDS layout (bytes):
//   [0)                feat_buf[2] : 2 * ROWS*WSTR*4   = 2 * 6400  = 12800
//   [12800)            w_buf[2]    : 2 * COLS*WSTR*4   = 2 * 81920 = 163840
//   [176640)           bias_s      : COLS*4            = 4096
//   [180736)           score_s     : 16*25*4           = 1600
//   [182336)           attn_s      : 16*25*4           = 1600
//   total 183936.  qv_s (ROWS*QV_STR*2 = 165120) aliases [0,165120) -- staging
//   buffers are dead once the K loop's final barrier has been passed.
#define FEAT_BYTES  (ROWS * WSTR * 4)        // 6400
#define WBUF_BYTES  (COLS * WSTR * 4)        // 81920
#define W_BASE      (2 * FEAT_BYTES)         // 12800
#define BIAS_BASE   (W_BASE + 2 * WBUF_BYTES)
#define SCORE_BASE  (BIAS_BASE + COLS * 4)
#define ATTN_BASE   (SCORE_BASE + BT * NMODELS * NMODELS * 4)
#define SMEM_TOTAL  (ATTN_BASE + BT * NMODELS * NMODELS * 4)

__device__ __forceinline__ float bf2f(unsigned int h16) {
    union { unsigned int u; float f; } c;
    c.u = h16 << 16;
    return c.f;
}

__device__ __forceinline__ unsigned int f2bf(float f) {
    union { float f; unsigned int u; } c;
    c.f = f;
    unsigned int u = c.u;
    u += 0x7fffu + ((u >> 16) & 1u);   // round-to-nearest-even
    return u >> 16;
}

// ---- pass 0: pack Wq||Wv fp32 -> bf16x2 dwords in workspace ---------------
__global__ void pack_weights_bf16(const float* __restrict__ Wq,
                                  const float* __restrict__ Wv,
                                  unsigned int* __restrict__ wqv) {
    int idx = blockIdx.x * blockDim.x + threadIdx.x;   // over COLS * IN_DIM/2
    if (idx >= COLS * (IN_DIM / 2)) return;
    int n  = idx / (IN_DIM / 2);
    int kd = idx % (IN_DIM / 2);
    const float* src = (n < QDIM) ? (Wq + (size_t)n * IN_DIM)
                                  : (Wv + (size_t)(n - QDIM) * IN_DIM);
    float a = src[kd * 2 + 0];
    float b = src[kd * 2 + 1];
    wqv[idx] = f2bf(a) | (f2bf(b) << 16);
}

// ---- fused kernel ----------------------------------------------------------
__global__ void __launch_bounds__(1024, 1)
fused_cross_attn(const float* __restrict__ feat,          // [M,B,D] fp32
                 const float* __restrict__ pv,            // [M,B] fp32
                 const unsigned int* __restrict__ wqv,    // packed bf16 weights or null
                 const float* __restrict__ Wq,            // fp32 fallback
                 const float* __restrict__ Wv,
                 const float* __restrict__ bq,
                 const float* __restrict__ bv,
                 float* __restrict__ out) {               // [B, M*QDIM] fp32
    extern __shared__ unsigned char smem[];
    unsigned short* qv_s    = (unsigned short*)smem;                 // aliases staging
    float*          bias_s  = (float*)(smem + BIAS_BASE);
    float*          score_s = (float*)(smem + SCORE_BASE);
    float*          attn_s  = (float*)(smem + ATTN_BASE);

    const int t      = threadIdx.x;
    const int lane   = t & 31;
    const int wave   = t >> 5;                 // 0..31
    const int half   = (lane < 16) ? 0 : 1;   // wave halves per WMMA layout
    const int ln     = lane & 15;
    const int wg_b0  = blockIdx.x * BT;

    // stage biases (bq || bv) -- survives the whole GEMM phase
    if (t < COLS) bias_s[t] = (t < QDIM) ? bq[t] : bv[t - QDIM];

    // 2 column tiles x 5 row tiles of 16x16 f32 accumulators per wave
    v8f acc[2][NMODELS];
#pragma unroll
    for (int c = 0; c < 2; ++c)
#pragma unroll
        for (int r = 0; r < NMODELS; ++r) acc[c][r] = v8f{};

    // ---- weight staging assignment: 4 async b128 per thread per chunk ----
    //   sub-op i: row n = t/4 + 256*i, dword group kd = (t%4)*4
    const int wn  = t >> 2;           // 0..255
    const int wkd = (t & 3) * 4;      // 0,4,8,12
    const unsigned int w_lane_off = (unsigned int)((wn * WSTR + wkd) * 4); // bytes in buf

    // ---- feature staging assignment: one float4 per thread (t < 640) ----
    const int frow = t / (KC / 4);    // 0..79
    const int ff4  = t % (KC / 4);    // 0..7
    const float* fsrc0 = feat
        + ((size_t)((frow / BT) * BATCH + wg_b0 + (frow % BT))) * IN_DIM + ff4 * 4;
    const unsigned int f_lane_off = (unsigned int)(frow * WSTR * 4 + ff4 * 8); // bytes

    // ---- fragment base pointers (buffer 0; buffer select adds a constant) ----
    //   A frag (16-bit A 16x32): row M = ln; uint4 blocks {half, half+2}
    //   B frag (16-bit B 32x16): col N = ln; uint4 blocks {half, half+2}
    const uint4* a_base = (const uint4*)smem + ln * (WSTR / 4);
    const uint4* b_base = (const uint4*)(smem + W_BASE) + (wave * 16 + ln) * (WSTR / 4);

    // ---- staging helpers ----
    auto stage_weights = [&](int kcc, int buf) {
        if (wqv) {
            unsigned int lds0 = W_BASE + buf * WBUF_BYTES + w_lane_off;
#pragma unroll
            for (int i = 0; i < 4; ++i) {
                unsigned long long gaddr = (unsigned long long)
                    (wqv + (size_t)(wn + 256 * i) * (IN_DIM / 2) + kcc * (KC / 2) + wkd);
                unsigned int lds_off = lds0 + i * 256 * WSTR * 4;
                asm volatile("global_load_async_to_lds_b128 %0, %1, off"
                             :: "v"(lds_off), "v"(gaddr) : "memory");
            }
        } else {
            unsigned int* wbuf = (unsigned int*)(smem + W_BASE + buf * WBUF_BYTES);
            for (int e = t; e < COLS * (KC / 2); e += 1024) {
                int n  = e / (KC / 2);
                int kd = e % (KC / 2);
                const float* src = ((n < QDIM) ? (Wq + (size_t)n * IN_DIM)
                                               : (Wv + (size_t)(n - QDIM) * IN_DIM))
                                   + kcc * KC + kd * 2;
                wbuf[n * WSTR + kd] = f2bf(src[0]) | (f2bf(src[1]) << 16);
            }
        }
    };
    auto store_features = [&](float4 v, int buf) {
        if (t < ROWS * (KC / 4)) {
            unsigned int d0 = f2bf(v.x) | (f2bf(v.y) << 16);
            unsigned int d1 = f2bf(v.z) | (f2bf(v.w) << 16);
            *(uint2*)(smem + buf * FEAT_BYTES + f_lane_off) = make_uint2(d0, d1);
        }
    };

    // ---- prologue: stage chunk 0 into buffer 0 ----
    stage_weights(0, 0);
    {
        float4 v = {};
        if (t < ROWS * (KC / 4)) v = *(const float4*)fsrc0;
        store_features(v, 0);
    }
    if (wqv) asm volatile("s_wait_asynccnt 0x0" ::: "memory");
    __syncthreads();

    // ------------------- pipelined main GEMM over K -------------------
    for (int kc = 0; kc < NKC; ++kc) {
        const int cur = kc & 1;
        const int nxt = cur ^ 1;
        const bool more = (kc + 1 < NKC);

        // kick off next chunk: weight DMA to LDS + feature load into registers
        float4 fv = {};
        if (more) {
            stage_weights(kc + 1, nxt);
            if (t < ROWS * (KC / 4))
                fv = *(const float4*)(fsrc0 + (kc + 1) * KC);
        }

        // ---- compute current chunk ----
        const uint4* ab = a_base + cur * (FEAT_BYTES / 16);
        const uint4* bb = b_base + cur * (WBUF_BYTES / 16);
        union { uint4 q[2]; v16bf v; } bf0, bf1;
        bf0.q[0] = bb[half + 0];
        bf0.q[1] = bb[half + 2];
        bf1.q[0] = bb[32 * 16 * (WSTR / 4) + half + 0];
        bf1.q[1] = bb[32 * 16 * (WSTR / 4) + half + 2];
#pragma unroll
        for (int r = 0; r < NMODELS; ++r) {
            union { uint4 q[2]; v16bf v; } af;     // A fragment: 2 x ds_load_b128
            af.q[0] = ab[r * 16 * (WSTR / 4) + half + 0];
            af.q[1] = ab[r * 16 * (WSTR / 4) + half + 2];
            acc[0][r] = __builtin_amdgcn_wmma_f32_16x16x32_bf16(
                false, af.v, false, bf0.v, (short)0, acc[0][r], false, false);
            acc[1][r] = __builtin_amdgcn_wmma_f32_16x16x32_bf16(
                false, af.v, false, bf1.v, (short)0, acc[1][r], false, false);
        }

        // ---- finish next-chunk staging, one barrier per iteration ----
        if (more) {
            store_features(fv, nxt);
            if (wqv) asm volatile("s_wait_asynccnt 0x0" ::: "memory");
        }
        __syncthreads();
    }

    // ------------------- write q,v tiles (+bias) to LDS as bf16 ----------------
    // (staging buffers are dead now; qv_s aliases them)
#pragma unroll
    for (int c = 0; c < 2; ++c) {
        int ctg  = wave + 32 * c;
        int ncol = ctg * 16 + ln;          // C/D layout: N = lane % 16
        float bias = bias_s[ncol];
#pragma unroll
        for (int r = 0; r < NMODELS; ++r) {
#pragma unroll
            for (int r8 = 0; r8 < 8; ++r8) {
                int mrow = r * 16 + r8 + half * 8;   // C/D layout: M = r8 + 8*half
                float val = acc[c][r][r8] + bias;
                qv_s[mrow * QV_STR + ncol] = (unsigned short)f2bf(val);
            }
        }
    }
    __syncthreads();

    // ------------------- scores: 16 b x 25 (i,j) dots over 512 dims ------------
    if (t < BT * NMODELS * NMODELS) {
        int b  = t / (NMODELS * NMODELS);
        int ij = t % (NMODELS * NMODELS);
        int i  = ij / NMODELS;
        int j  = ij % NMODELS;
        const unsigned int* ua = (const unsigned int*)(qv_s + (i * 16 + b) * QV_STR);
        const unsigned int* ub = (const unsigned int*)(qv_s + (j * 16 + b) * QV_STR);
        float s = 0.f;
#pragma unroll 8
        for (int d = 0; d < QDIM / 2; ++d) {
            unsigned int x = ua[d], y = ub[d];
            s += bf2f(x & 0xffffu) * bf2f(y & 0xffffu)
               + bf2f(x >> 16)     * bf2f(y >> 16);
        }
        float pvi = pv[(size_t)i * BATCH + wg_b0 + b];
        float pvj = pv[(size_t)j * BATCH + wg_b0 + b];
        score_s[t] = s - UW * (pvi + pvj);
    }
    __syncthreads();

    // ------------------- softmax over j (rows of 5) -------------------
    if (t < BT * NMODELS) {
        int b = t / NMODELS;
        int i = t % NMODELS;
        const float* srow = score_s + b * 25 + i * 5;
        float mx = srow[0];
#pragma unroll
        for (int j = 1; j < NMODELS; ++j) mx = fmaxf(mx, srow[j]);
        float e[NMODELS], sum = 0.f;
#pragma unroll
        for (int j = 0; j < NMODELS; ++j) { e[j] = __expf(srow[j] - mx); sum += e[j]; }
        float inv = 1.f / sum;
#pragma unroll
        for (int j = 0; j < NMODELS; ++j) attn_s[b * 25 + i * 5 + j] = e[j] * inv;
    }
    __syncthreads();

    // ------------------- z = attn @ v ; coalesced float2 stores -------------------
    for (int e2 = t; e2 < BT * NMODELS * (QDIM / 2); e2 += 1024) {
        int qd2 = e2 % (QDIM / 2);
        int rem = e2 / (QDIM / 2);
        int i   = rem % NMODELS;
        int b   = rem / NMODELS;
        float a0 = attn_s[b * 25 + i * 5 + 0];
        float a1 = attn_s[b * 25 + i * 5 + 1];
        float a2 = attn_s[b * 25 + i * 5 + 2];
        float a3 = attn_s[b * 25 + i * 5 + 3];
        float a4 = attn_s[b * 25 + i * 5 + 4];
        float z0 = 0.f, z1 = 0.f;
#pragma unroll
        for (int j = 0; j < NMODELS; ++j) {
            const unsigned int* vr = (const unsigned int*)(qv_s + (j * 16 + b) * QV_STR);
            unsigned int x = vr[QDIM / 2 + qd2];   // V region: cols 512..1023
            float aj = (j == 0) ? a0 : (j == 1) ? a1 : (j == 2) ? a2 : (j == 3) ? a3 : a4;
            z0 += aj * bf2f(x & 0xffffu);
            z1 += aj * bf2f(x >> 16);
        }
        size_t off = (size_t)(wg_b0 + b) * (NMODELS * QDIM) + (size_t)i * QDIM + qd2 * 2;
        *(float2*)(out + off) = make_float2(z0, z1);
    }
}

// ---------------------------------------------------------------------------
extern "C" void kernel_launch(void* const* d_in, const int* in_sizes, int n_in,
                              void* d_out, int out_size, void* d_ws, size_t ws_size,
                              hipStream_t stream) {
    const float* feat = (const float*)d_in[0];   // [5,16384,1024]
    const float* pv   = (const float*)d_in[1];   // [5,16384]
    const float* Wq   = (const float*)d_in[2];   // [512,1024]
    const float* bq   = (const float*)d_in[3];   // [512]
    // d_in[4] = Wk, d_in[5] = bk : dead inputs (reference sets k = q)
    const float* Wv   = (const float*)d_in[6];   // [512,1024]
    const float* bv   = (const float*)d_in[7];   // [512]
    float* out = (float*)d_out;

    (void)in_sizes; (void)n_in; (void)out_size;

    unsigned int* wqv = nullptr;
    const size_t wbytes = (size_t)COLS * (IN_DIM / 2) * sizeof(unsigned int); // 2 MB
    if (ws_size >= wbytes) {
        wqv = (unsigned int*)d_ws;
        int total = COLS * (IN_DIM / 2);
        pack_weights_bf16<<<(total + 255) / 256, 256, 0, stream>>>(Wq, Wv, wqv);
    }

    size_t shmem = SMEM_TOTAL;   // ~184 KB (< 320 KB per WGP)
    fused_cross_attn<<<BATCH / BT, 1024, shmem, stream>>>(
        feat, pv, wqv, Wq, Wv, bq, bv, out);
}
